// LTriple_5720896438537
// MI455X (gfx1250) — matrix operand order repaired and checked by reading, not compiled, not run on hardware
//
#include <hip/hip_runtime.h>

// ---- MI455X / gfx1250: wave32, WMMA bf16 16x16x32, f32 accumulate ----
typedef __attribute__((ext_vector_type(16))) __bf16 v16bf;
typedef __attribute__((ext_vector_type(8)))  __bf16 v8bf;
typedef __attribute__((ext_vector_type(8)))  float  v8f;
typedef __attribute__((ext_vector_type(4)))  float  v4f;

#define B_  8
#define N_  8192
#define D_  128
#define S_  8
// GEMM1: M-tile 16 x (K=384 -> 12 ktiles) x (N=256 -> 16 ntiles)
// GEMM2: M-tile 16 x (K=256 ->  8 ktiles) x (N=128 ->  8 otiles)
#define W1_FRAG_ELEMS (12 * 16 * 512)   // 98304 bf16
#define W2_FRAG_ELEMS (8 * 8 * 512)     // 32768 bf16

#define PF1 8   // B-fragment pipeline depth, GEMM1 (8 frags = 16 b128 in flight)
#define PF2 8   // B-fragment pipeline depth, GEMM2

// Load one 32x16 bf16 B-fragment (32 bytes per lane) from a per-lane pointer.
__device__ __forceinline__ v16bf load_frag(const __bf16* lp) {
  v8bf lo = *(const v8bf*)lp;
  v8bf hi = *(const v8bf*)(lp + 8);
  v16bf r;
#pragma unroll
  for (int e = 0; e < 8; ++e) { r[e] = lo[e]; r[8 + e] = hi[e]; }
  return r;
}

// ---------------------------------------------------------------------------
// Prep: convert W1/W2 (f32 row-major) into bf16 fragments in WMMA B-matrix
// (32x16) shape, in this file's contiguous k order: lane l holds column N = l%16,
// K = (l>=16 ? 16 : 0) + e, e = 0..15, 16 consecutive bf16 per lane.
// ---------------------------------------------------------------------------
__global__ void prep_weights(const float* __restrict__ W1,
                             const float* __restrict__ W2,
                             __bf16* __restrict__ w1f,
                             __bf16* __restrict__ w2f) {
  int t = blockIdx.x * 256 + threadIdx.x;
  if (t < W1_FRAG_ELEMS) {
    int e    = t & 15;
    int lane = (t >> 4) & 31;
    int frag = t >> 9;
    int kt   = frag >> 4;          // 0..11
    int nt   = frag & 15;          // 0..15
    int K    = kt * 32 + (lane >> 4) * 16 + e;
    int Nc   = nt * 16 + (lane & 15);
    w1f[t] = (__bf16)W1[K * 256 + Nc];
  }
  if (t < W2_FRAG_ELEMS) {
    int e    = t & 15;
    int lane = (t >> 4) & 31;
    int frag = t >> 9;
    int kt   = frag >> 3;          // 0..7
    int ot   = frag & 7;           // 0..7
    int K    = kt * 32 + (lane >> 4) * 16 + e;
    int Nc   = ot * 16 + (lane & 15);
    w2f[t] = (__bf16)W2[K * 128 + Nc];
  }
}

// ---------------------------------------------------------------------------
// Main kernel. Each wave handles 2 n's x 8 s = a 16-row M-tile end to end.
// Block = 8 waves = 16 n's. No cross-wave communication, no barriers.
// ---------------------------------------------------------------------------
__global__ __launch_bounds__(256, 1)
void triple_mlp_kernel(const float* __restrict__ x,
                       const int*   __restrict__ ji,
                       const int*   __restrict__ ki,
                       const __bf16* __restrict__ w1f,
                       const __bf16* __restrict__ w2f,
                       const float* __restrict__ b1,
                       const float* __restrict__ b2,
                       float* __restrict__ out) {
  // Per-wave 16x256 bf16 GELU-activation staging (lane-transpose for GEMM2).
  __shared__ __align__(16) __bf16 Hs[8][16][256];   // 64 KB

  const int lane = threadIdx.x & 31;
  const int wave = threadIdx.x >> 5;
  const int gn0  = blockIdx.x * 16 + wave * 2;   // first of this wave's 2 flat n's
  const int m    = lane & 15;                    // A-layout: row M = lane%16
  const int hi   = lane >> 4;                    // lane half selects K sub-stripe
  const int s    = m & 7;                        // rows 0..7 -> n0, 8..15 -> n1
  const int gn   = gn0 + (m >> 3);               // this lane's flat (b,n)
  const int b    = gn >> 13;                     // N_ = 8192

  // Per-lane fragment base pointers (fragment index becomes an imm offset).
  const __bf16* w1l = w1f + lane * 16;
  const __bf16* w2l = w2f + lane * 16;

  // Preload per-lane bias values so no scalar VMEM loads land mid-pipeline.
  float b1v[16], b2v[8];
#pragma unroll
  for (int nt = 0; nt < 16; ++nt) b1v[nt] = b1[nt * 16 + m];
#pragma unroll
  for (int ot = 0; ot < 8; ++ot)  b2v[ot] = b2[ot * 16 + m];

  // Gather the three source rows for this lane's M row.
  const int jidx = ji[gn * S_ + s];
  const int kidx = ki[gn * S_ + s];
  const float* rowi = x + (size_t)gn * D_;
  const float* rowj = x + ((size_t)(b << 13) + jidx) * D_;
  const float* rowk = x + ((size_t)(b << 13) + kidx) * D_;
  const float* rows[3] = { rowi, rowj, rowk };

  // ---- Build all 12 A-fragments (16x32 bf16) in registers: 96 VGPRs.
  // A k order used here (same as B prep): lane's 16 values are K = koff+0..15,
  // koff = hi*16, within the 32-wide K tile.
  v16bf A1[12];
#pragma unroll
  for (int kt = 0; kt < 12; ++kt) {
    const float* r = rows[kt >> 2];              // xi | xj | xk segment
    const int base = (kt & 3) * 32 + hi * 16;
    v4f f0 = *(const v4f*)(r + base);
    v4f f1 = *(const v4f*)(r + base + 4);
    v4f f2 = *(const v4f*)(r + base + 8);
    v4f f3 = *(const v4f*)(r + base + 12);
#pragma unroll
    for (int e = 0; e < 4; ++e) {
      A1[kt][e]      = (__bf16)f0[e];
      A1[kt][4 + e]  = (__bf16)f1[e];
      A1[kt][8 + e]  = (__bf16)f2[e];
      A1[kt][12 + e] = (__bf16)f3[e];
    }
  }

  // ---- GEMM1: flat software-pipelined stream of 16x12 = 192 B fragments.
  // Pipeline index g = nt*12 + kt maps to memory fragment (kt*16 + nt).
  v16bf Bbuf[PF1];
#pragma unroll
  for (int i = 0; i < PF1; ++i) {
    const int nt0 = i / 12, kt0 = i % 12;
    Bbuf[i] = load_frag(w1l + (kt0 * 16 + nt0) * 512);
  }

#pragma unroll
  for (int nt = 0; nt < 16; ++nt) {
    const float bias = b1v[nt];
    v8f acc = { bias, bias, bias, bias, bias, bias, bias, bias };
#pragma unroll
    for (int kt = 0; kt < 12; ++kt) {
      const int g = nt * 12 + kt;
      acc = __builtin_amdgcn_wmma_f32_16x16x32_bf16(
          false, A1[kt], false, Bbuf[g % PF1], (short)0, acc, false, false);
      const int gp = g + PF1;                      // refill this slot
      if (gp < 192) {
        const int ntp = gp / 12, ktp = gp % 12;
        Bbuf[g % PF1] = load_frag(w1l + (ktp * 16 + ntp) * 512);
      }
    }
    // GELU (sigmoid form) + bf16 convert, store to LDS.
    // C-layout: VGPR r holds row M = r + hi*8, column N = nt*16 + m.
#pragma unroll
    for (int r = 0; r < 8; ++r) {
      const float v = acc[r];
      const float g = v * __builtin_amdgcn_rcpf(1.0f + __expf(-1.702f * v));
      Hs[wave][r + hi * 8][nt * 16 + m] = (__bf16)g;
    }
  }

  // ---- Reload activations from LDS in A-fragment layout (lane transpose).
  v16bf A2[8];
#pragma unroll
  for (int kt = 0; kt < 8; ++kt) {
    const int base = kt * 32 + hi * 16;
    const __bf16* hp = &Hs[wave][m][0];
    v8bf h0 = *(const v8bf*)(hp + base);
    v8bf h1 = *(const v8bf*)(hp + base + 8);
#pragma unroll
    for (int e = 0; e < 8; ++e) { A2[kt][e] = h0[e]; A2[kt][8 + e] = h1[e]; }
  }

  // ---- GEMM2: flat pipelined stream of 8x8 = 64 B fragments.
  // Pipeline index g = ot*8 + kt maps to memory fragment (kt*8 + ot).
  v16bf B2buf[PF2];
#pragma unroll
  for (int i = 0; i < PF2; ++i) {
    const int ot0 = i / 8, kt0 = i % 8;
    B2buf[i] = load_frag(w2l + (kt0 * 8 + ot0) * 512);
  }

#pragma unroll
  for (int ot = 0; ot < 8; ++ot) {
    const float bias = b2v[ot];
    // Init each VGPR with b2 => after the s-mean (sum * 1/8), b2 appears once.
    v8f acc = { bias, bias, bias, bias, bias, bias, bias, bias };
#pragma unroll
    for (int kt = 0; kt < 8; ++kt) {
      const int g = ot * 8 + kt;
      acc = __builtin_amdgcn_wmma_f32_16x16x32_bf16(
          false, A2[kt], false, B2buf[g % PF2], (short)0, acc, false, false);
      const int gp = g + PF2;
      if (gp < 64) {
        const int otp = gp / 8, ktp = gp % 8;
        B2buf[g % PF2] = load_frag(w2l + (ktp * 8 + otp) * 512);
      }
    }
    // S-mean falls out of the C layout: VGPR r = row s=r (lanes 0-15 -> n0)
    // and s=r+8's n1 rows for lanes 16-31. Sum the 8 VGPRs, scale by 1/8.
    const float sum = ((acc[0] + acc[1]) + (acc[2] + acc[3])) +
                      ((acc[4] + acc[5]) + (acc[6] + acc[7]));
    out[(size_t)(gn0 + hi) * D_ + ot * 16 + m] = sum * 0.125f;
  }
}

// ---------------------------------------------------------------------------
extern "C" void kernel_launch(void* const* d_in, const int* in_sizes, int n_in,
                              void* d_out, int out_size, void* d_ws, size_t ws_size,
                              hipStream_t stream) {
  const float* x  = (const float*)d_in[0];
  const int*   ji = (const int*)d_in[1];
  const int*   ki = (const int*)d_in[2];
  const float* W1 = (const float*)d_in[3];
  const float* b1 = (const float*)d_in[4];
  const float* W2 = (const float*)d_in[5];
  const float* b2 = (const float*)d_in[6];

  __bf16* w1f = (__bf16*)d_ws;
  __bf16* w2f = (__bf16*)((char*)d_ws + (size_t)W1_FRAG_ELEMS * sizeof(__bf16));

  prep_weights<<<(W1_FRAG_ELEMS + 255) / 256, 256, 0, stream>>>(W1, W2, w1f, w2f);

  // 65536 flat n's / 16 per block = 4096 blocks of 256 threads (8 waves).
  triple_mlp_kernel<<<(B_ * N_) / 16, 256, 0, stream>>>(
      x, ji, ki, w1f, w2f, b1, b2, (float*)d_out);
}
